// MultiHeadAttentionEQ_40355512713878
// MI455X (gfx1250) — matrix operand order repaired and checked by reading, not compiled, not run on hardware
//
#include <hip/hip_runtime.h>
#include <hip/hip_bf16.h>
#include <math.h>

// ---------------------------------------------------------------------------
// Types for CDNA5 WMMA (wave32): v_wmma_f32_16x16x32_bf16
// ---------------------------------------------------------------------------
typedef __attribute__((ext_vector_type(8)))  __bf16 v8bf;
typedef __attribute__((ext_vector_type(16))) __bf16 v16bf;
typedef __attribute__((ext_vector_type(8)))  float  v8f;

#define WMMA_BF16(acc, a, b) \
  __builtin_amdgcn_wmma_f32_16x16x32_bf16(false, (a), false, (b), (short)0, (acc), false, false)

// f32 -> bf16: native cast (backend emits v_cvt_pk_bf16_f32 for pairs)
__device__ __forceinline__ __bf16 f2bf(float f) { return (__bf16)f; }

// ---------------------------------------------------------------------------
// Fragment loaders per CDNA5 ISA layouts (16-bit A 16x32, 16-bit B 32x16).
// A: lane L holds row M = L%16; K = kbase + (L/16)*8 + {0..7} and +16+{0..7}
// B: lane L holds col N = L%16; K = kbase + (L/16)*8 + {0..7} and +16+{0..7}
// ---------------------------------------------------------------------------
__device__ __forceinline__ v16bf load_afrag_bf16(const __bf16* base, int row,
                                                 int stride, int kbase, int lane) {
  const __bf16* p = base + row * stride + kbase + ((lane >> 4) << 3);
  v8bf lo = *(const v8bf*)(p);
  v8bf hi = *(const v8bf*)(p + 16);
  return __builtin_shufflevector(lo, hi, 0,1,2,3,4,5,6,7,8,9,10,11,12,13,14,15);
}

__device__ __forceinline__ v16bf load_bfrag_bf16(const __bf16* base, int row,
                                                 int stride, int kbase, int lane) {
  const __bf16* p = base + row * stride + kbase + ((lane >> 4) << 3);
  v8bf lo = *(const v8bf*)(p);
  v8bf hi = *(const v8bf*)(p + 16);
  return __builtin_shufflevector(lo, hi, 0,1,2,3,4,5,6,7,8,9,10,11,12,13,14,15);
}

__device__ __forceinline__ v16bf load_afrag_f32(const float* base, int row,
                                                int stride, int kbase, int lane) {
  const float* p = base + row * stride + kbase + ((lane >> 4) << 3);
  v16bf r;
#pragma unroll
  for (int i = 0; i < 8; i++) r[i] = f2bf(p[i]);
#pragma unroll
  for (int i = 0; i < 8; i++) r[8 + i] = f2bf(p[16 + i]);
  return r;
}

__device__ __forceinline__ v16bf load_bfrag_f32(const float* base, int row,
                                                int stride, int kbase, int lane) {
  const float* p = base + row * stride + kbase + ((lane >> 4) << 3);
  v16bf r;
#pragma unroll
  for (int i = 0; i < 16; i++) r[i] = f2bf(p[i < 8 ? i : i + 8]);
  return r;
}

// CDNA5 async copy: 16B global -> LDS per lane, tracked by ASYNCcnt.
__device__ __forceinline__ void async_copy_b128(unsigned lds_byte_addr,
                                                const void* gaddr) {
  asm volatile("global_load_async_to_lds_b128 %0, %1, off"
               :: "v"(lds_byte_addr), "v"(gaddr) : "memory");
}
__device__ __forceinline__ void wait_async0() {
  asm volatile("s_wait_asynccnt 0x0" ::: "memory");
}

// ---------------------------------------------------------------------------
// Constants:  B=1 A=4 N=M=1024 D=256 H=4 C=64
// ---------------------------------------------------------------------------
#define NROWS 1024
#define DDIM  256
#define HEADS 4
#define CDIM  64
#define ANCH  4
#define QSZ   (ANCH * NROWS * DDIM)   // elements of one bf16 activation buffer

// ===========================================================================
// Kernel 1: projection  out = X @ W^T + b   (X:[4096,256]f32, W:[256,256]f32)
// transpose_out==0 -> out[n][o] bf16 row-major (rowlen 256)
// transpose_out==1 -> VT[e][o][m] bf16 (e=n/1024, m=n%1024, row stride 1024)
// ===========================================================================
__global__ void k_proj(const float* __restrict__ X, const float* __restrict__ W,
                       const float* __restrict__ bias, __bf16* __restrict__ out,
                       int transpose_out) {
  const int tid = threadIdx.x, lane = tid & 31, w = tid >> 5;
  const int tile = blockIdx.x * 8 + w;          // 4096 tiles (256 x 16)
  const int n0 = (tile >> 4) << 4;              // row-tile * 16
  const int o0 = (tile & 15) << 4;              // col-tile * 16

  v8f acc = {};
#pragma unroll
  for (int kc = 0; kc < 8; kc++) {
    v16bf a = load_afrag_f32(X, n0 + (lane & 15), DDIM, kc * 32, lane);
    v16bf b = load_bfrag_f32(W, o0 + (lane & 15), DDIM, kc * 32, lane);
    acc = WMMA_BF16(acc, a, b);
  }
  const float bv = bias[o0 + (lane & 15)];

  if (!transpose_out) {
#pragma unroll
    for (int r = 0; r < 8; r++) {
      int n = n0 + r + ((lane >> 4) << 3);
      out[n * DDIM + o0 + (lane & 15)] = f2bf(acc[r] + bv);
    }
  } else {
    // pack 8 consecutive m into one 16B store
    v8bf pk;
#pragma unroll
    for (int r = 0; r < 8; r++) pk[r] = f2bf(acc[r] + bv);
    int e = n0 >> 10;
    int m = (n0 & 1023) + ((lane >> 4) << 3);
    int o = o0 + (lane & 15);
    *(v8bf*)(out + (size_t)e * (DDIM * NROWS) + o * NROWS + m) = pk;
  }
}

// ===========================================================================
// Kernel 2a: zero the (a,e) statistics
// ===========================================================================
__global__ void k_zero(float* __restrict__ G) {
  if (threadIdx.x < 16) G[threadIdx.x] = 0.0f;
}

// ===========================================================================
// Kernel 2b: anchor stats  G[a,e] += sum_{n,m} (Q_a[n,:] . K_e[m,:])^2
// Q/K 64x256 panels staged into LDS via global_load_async_to_lds_b128.
// grid: x = 256 (64x64 tiles over 1024x1024), y = 16 (a*4+e)
// LDS: QL[64][264]bf16 (33792B) + KL[64][264]bf16 + red[256]f32
// ===========================================================================
#define ST_STRIDE   264                      // 256 + 8 pad (132 dw ≡ 4 mod 64)
#define ST_MAT_B    (64 * ST_STRIDE * 2)     // 33792
#define ST_RED_OFF  (2 * ST_MAT_B)           // 67584
#define ST_SMEM     (ST_RED_OFF + 256 * 4)   // 68608

__global__ void k_stats(const __bf16* __restrict__ Qb, const __bf16* __restrict__ Kb,
                        float* __restrict__ G) {
  extern __shared__ char sm[];
  __bf16* QL  = (__bf16*)sm;
  __bf16* KL  = (__bf16*)(sm + ST_MAT_B);
  float*  red = (float*)(sm + ST_RED_OFF);

  const int tid = threadIdx.x, lane = tid & 31, w = tid >> 5;
  const int a = blockIdx.y >> 2, e = blockIdx.y & 3;
  const int tn = (blockIdx.x >> 4) * 64, tm = (blockIdx.x & 15) * 64;
  const __bf16* Qg = Qb + (size_t)a * (NROWS * DDIM) + (size_t)tn * DDIM;
  const __bf16* Kg = Kb + (size_t)e * (NROWS * DDIM) + (size_t)tm * DDIM;

  // ---- async stage: 64 rows x 256 bf16 (32 x 16B chunks per row) ----------
#pragma unroll
  for (int i = 0; i < 8; i++) {
    int t = i * 256 + tid;                   // 0..2047 chunk id per matrix
    int row = t >> 5, ch = (t & 31) << 4;    // ch = byte offset within row
    async_copy_b128((unsigned)(uintptr_t)(QL + row * ST_STRIDE) + ch,
                    (const char*)(Qg + row * DDIM) + ch);
    async_copy_b128((unsigned)(uintptr_t)(KL + row * ST_STRIDE) + ch,
                    (const char*)(Kg + row * DDIM) + ch);
  }
  wait_async0();
  __syncthreads();

  // ---- WMMA on staged panels ----------------------------------------------
  float local = 0.0f;
#pragma unroll
  for (int i = 0; i < 2; i++) {
    int t = w * 2 + i;
    int n0 = (t >> 2) << 4;                  // local rows in QL
    int m0 = (t & 3) << 4;                   // local rows in KL
    v8f acc = {};
#pragma unroll
    for (int kc = 0; kc < 8; kc++) {
      v16bf af = load_afrag_bf16(QL, n0 + (lane & 15), ST_STRIDE, kc * 32, lane);
      v16bf bf = load_bfrag_bf16(KL, m0 + (lane & 15), ST_STRIDE, kc * 32, lane);
      acc = WMMA_BF16(acc, af, bf);
    }
#pragma unroll
    for (int r = 0; r < 8; r++) local += acc[r] * acc[r];
  }

  red[tid] = local;
  __syncthreads();
  for (int s = 128; s > 0; s >>= 1) {
    if (tid < s) red[tid] += red[tid + s];
    __syncthreads();
  }
  if (tid == 0) atomicAdd(&G[blockIdx.y], red[0]);
}

// ===========================================================================
// Kernel 2c: attn_ae[a,e] = G[a,e] / sum_e G[a,e]
// ===========================================================================
__global__ void k_norm(const float* __restrict__ G, float* __restrict__ AE) {
  int t = threadIdx.x;
  if (t < 16) {
    int a = t >> 2;
    float s = G[a * 4 + 0] + G[a * 4 + 1] + G[a * 4 + 2] + G[a * 4 + 3];
    AE[t] = G[t] / s;
  }
}

// ===========================================================================
// Kernel 3: attention.  grid: x = 32 (n-tiles of 32), y = 16 (a*4+h).
// 256 threads (8 waves). Dynamic LDS:
//   S : f32  [32][1028]  (raw scores, one key-anchor at a time)
//   P : bf16 [32][1032]  (weighted probabilities)
//   red[256], rowmax[32], rowmul[32]
// ===========================================================================
#define SS 1028
#define PS 1032
#define S_BYTES   (32 * SS * 4)                 // 131584
#define P_BYTES   (32 * PS * 2)                 // 66048
#define RED_OFF   (S_BYTES + P_BYTES)           // 197632
#define RMAX_OFF  (RED_OFF + 256 * 4)           // 198656
#define RMUL_OFF  (RMAX_OFF + 32 * 4)           // 198784
#define SMEM_BYTES (RMUL_OFF + 32 * 4)          // 198912

__global__ void k_attn(const __bf16* __restrict__ Qb, const __bf16* __restrict__ Kb,
                       const __bf16* __restrict__ VT, const float* __restrict__ AE,
                       float* __restrict__ outp) {
  extern __shared__ char smem[];
  float*  S      = (float*)smem;
  __bf16* P      = (__bf16*)(smem + S_BYTES);
  float*  red    = (float*)(smem + RED_OFF);
  float*  rowmax = (float*)(smem + RMAX_OFF);
  float*  rowmul = (float*)(smem + RMUL_OFF);

  const int tid = threadIdx.x, lane = tid & 31, w = tid >> 5;
  const int a = blockIdx.y >> 2, h = blockIdx.y & 3;
  const int nb = blockIdx.x * 32;
  const int nsub = w & 1, csub = w >> 1;
  const float SCALE = 0.125f;                 // 1/sqrt(C)

  const __bf16* Qa = Qb + (size_t)a * (NROWS * DDIM);

  // Preload Q A-fragments for this wave's 16-row sub-tile (k chunks c=0..63)
  v16bf qf[2];
#pragma unroll
  for (int kc = 0; kc < 2; kc++)
    qf[kc] = load_afrag_bf16(Qa, nb + nsub * 16 + (lane & 15), DDIM,
                             h * CDIM + kc * 32, lane);

  v8f hacc = {};                              // output accumulator across e, m

  for (int e = 0; e < ANCH; e++) {
    const __bf16* Ke = Kb + (size_t)e * (NROWS * DDIM);
    const __bf16* Ve = VT + (size_t)e * (DDIM * NROWS);
    const float w_ae = AE[a * 4 + e];

    // ---- scores: S[0:32][0:1024] = Q_tile . K_e^T (head h slice) ----------
    for (int j = 0; j < 16; j++) {
      int mt = (w >> 1) + 4 * j;              // m-tile 0..63
      int m0 = mt << 4;
      if (j < 15)                             // prefetch next K tile rows
        __builtin_prefetch(Ke + (size_t)(m0 + 64 + (lane & 15)) * DDIM +
                           h * CDIM, 0, 3);
      v8f acc = {};
#pragma unroll
      for (int kc = 0; kc < 2; kc++) {
        v16bf bf = load_bfrag_bf16(Ke, m0 + (lane & 15), DDIM,
                                   h * CDIM + kc * 32, lane);
        acc = WMMA_BF16(acc, qf[kc], bf);
      }
#pragma unroll
      for (int r = 0; r < 8; r++) {
        int nl = nsub * 16 + r + ((lane >> 4) << 3);
        S[nl * SS + m0 + (lane & 15)] = acc[r];
      }
    }
    __syncthreads();

    // ---- softmax pass 1: row max (8 threads per row, 128 elems each) ------
    {
      int row = tid >> 3, part = tid & 7;
      const float* sp = S + row * SS + part * 128;
      float mx = -3.0e38f;
#pragma unroll 4
      for (int i = 0; i < 128; i++) mx = fmaxf(mx, sp[i]);
      red[tid] = mx;
      __syncthreads();
      if (part == 0) {
        float m2 = red[tid];
#pragma unroll
        for (int k = 1; k < 8; k++) m2 = fmaxf(m2, red[tid + k]);
        rowmax[row] = m2;
      }
      __syncthreads();

      // ---- pass 2: row sum of exp; fold attn_ae/rowsum into multiplier ----
      float mxr = rowmax[row];
      float sum = 0.0f;
#pragma unroll 4
      for (int i = 0; i < 128; i++) sum += __expf((sp[i] - mxr) * SCALE);
      red[tid] = sum;
      __syncthreads();
      if (part == 0) {
        float s2 = 0.0f;
#pragma unroll
        for (int k = 0; k < 8; k++) s2 += red[tid + k];
        rowmul[row] = w_ae / s2;
      }
      __syncthreads();

      // ---- pass 3: P = exp((s-max)*scale) * (w_ae/rowsum)  (bf16) ---------
      float ml = rowmul[row];
      __bf16* pp = P + row * PS + part * 128;
#pragma unroll 4
      for (int i = 0; i < 128; i++)
        pp[i] = f2bf(__expf((sp[i] - mxr) * SCALE) * ml);
    }
    __syncthreads();

    // ---- PV: hacc += P(32x1024) . V_e(1024x64), this wave's 16x16 tile ----
    for (int mc = 0; mc < 32; mc++) {
      if (mc < 31)                            // prefetch next V fragment row
        __builtin_prefetch(Ve + (size_t)(h * CDIM + csub * 16 + (lane & 15)) *
                           NROWS + (mc + 1) * 32, 0, 3);
      v16bf pa = load_afrag_bf16(P, nsub * 16 + (lane & 15), PS, mc * 32, lane);
      v16bf vb = load_bfrag_bf16(Ve, h * CDIM + csub * 16 + (lane & 15), NROWS,
                                 mc * 32, lane);
      hacc = WMMA_BF16(hacc, pa, vb);
    }
    __syncthreads();                          // P reused next iteration
  }

  // ---- store hidden -> out[b=0][a][n][h*64+c] (f32) -----------------------
#pragma unroll
  for (int r = 0; r < 8; r++) {
    int n = nb + nsub * 16 + r + ((lane >> 4) << 3);
    int d = h * CDIM + csub * 16 + (lane & 15);
    outp[((size_t)a * NROWS + n) * DDIM + d] = hacc[r];
  }
}

// ===========================================================================
// Launch
// ===========================================================================
extern "C" void kernel_launch(void* const* d_in, const int* in_sizes, int n_in,
                              void* d_out, int out_size, void* d_ws, size_t ws_size,
                              hipStream_t stream) {
  (void)in_sizes; (void)n_in; (void)out_size; (void)ws_size;
  const float* Xq = (const float*)d_in[0];
  const float* Xk = (const float*)d_in[1];
  const float* Xv = (const float*)d_in[2];
  const float* Wq = (const float*)d_in[3];
  const float* bq = (const float*)d_in[4];
  const float* Wk = (const float*)d_in[5];
  const float* bk = (const float*)d_in[6];
  const float* Wv = (const float*)d_in[7];
  const float* bv = (const float*)d_in[8];

  __bf16* Qb = (__bf16*)d_ws;          // [4][1024][256] bf16
  __bf16* Kb = Qb + QSZ;               // [4][1024][256] bf16
  __bf16* VT = Kb + QSZ;               // [4][256][1024] bf16 (transposed V)
  float*  G  = (float*)(VT + QSZ);     // [16]
  float*  AE = G + 16;                 // [16]

  k_proj<<<512, 256, 0, stream>>>(Xq, Wq, bq, Qb, 0);
  k_proj<<<512, 256, 0, stream>>>(Xk, Wk, bk, Kb, 0);
  k_proj<<<512, 256, 0, stream>>>(Xv, Wv, bv, VT, 1);

  k_zero<<<1, 32, 0, stream>>>(G);
  k_stats<<<dim3(256, 16), 256, ST_SMEM, stream>>>(Qb, Kb, G);
  k_norm<<<1, 16, 0, stream>>>(G, AE);

  k_attn<<<dim3(32, 16), 256, SMEM_BYTES, stream>>>(Qb, Kb, VT, AE, (float*)d_out);
}
